// PillarAttention_7215545057625
// MI455X (gfx1250) — matrix operand order repaired and checked by reading, not compiled, not run on hardware
//
#include <hip/hip_runtime.h>
#include <hip/hip_bf16.h>

// ---------------------------------------------------------------------------
// Types for CDNA5 WMMA (wave32): D(16x16,f32) = A(16x32,f16) x B(32x16,f16) + C
// ---------------------------------------------------------------------------
typedef __attribute__((ext_vector_type(16))) _Float16 v16h;
typedef __attribute__((ext_vector_type(8)))  _Float16 v8h;
typedef __attribute__((ext_vector_type(8)))  float    v8f;

union FragH { v16h v; v8h h[2]; };

#define BB 4
#define SS 2048
#define CC 256
#define HH 8
#define DD 32
#define FF 512

__device__ __forceinline__ v8f wmma16(v16h a, v16h b, v8f c) {
  // (neg_a, A, neg_b, B, c_mod, C, reuse_a, reuse_b)
  return __builtin_amdgcn_wmma_f32_16x16x32_f16(false, a, false, b, (short)0, c,
                                                false, false);
}

// A-fragment (16 rows x 32 k), f16, from row-major [rows, ld]:
//   lane = 16*half + r holds row (row0+r), k chunks [8*half..+7], [16+8*half..+7]
__device__ __forceinline__ v16h load_frag_a(const _Float16* __restrict__ p,
                                            int ld, int row0, int k0) {
  int lane = threadIdx.x & 31;
  int half = lane >> 4, r = lane & 15;
  const _Float16* base = p + (size_t)(row0 + r) * ld + (k0 + 8 * half);
  FragH f;
  f.h[0] = *(const v8h*)(base);
  f.h[1] = *(const v8h*)(base + 16);
  return f.v;
}

// B-fragment (32 k x 16 cols), f16; memory holds B column-major as W[col][k]
// (row-major [cols, ld] along k): lane = 16*half + n holds col (col0+n),
// k chunks [k0+8*half..+7], [k0+16+8*half..+7].
__device__ __forceinline__ v16h load_frag_b(const _Float16* __restrict__ p,
                                            int ld, int col0, int k0) {
  int lane = threadIdx.x & 31;
  int half = lane >> 4, n = lane & 15;
  const _Float16* base = p + (size_t)(col0 + n) * ld + (k0 + 8 * half);
  FragH f;
  f.h[0] = *(const v8h*)(base);
  f.h[1] = *(const v8h*)(base + 16);
  return f.v;
}

// ---------------------------------------------------------------------------
// fp32 -> fp16 conversion (4 elems / thread)
// ---------------------------------------------------------------------------
__global__ void cvt_f32_f16(const float* __restrict__ in,
                            _Float16* __restrict__ out, int n4) {
  int i = blockIdx.x * blockDim.x + threadIdx.x;
  if (i < n4) {
    float4 v = ((const float4*)in)[i];
    _Float16* o = out + (size_t)i * 4;
    o[0] = (_Float16)v.x; o[1] = (_Float16)v.y;
    o[2] = (_Float16)v.z; o[3] = (_Float16)v.w;
  }
}

// ---------------------------------------------------------------------------
// QKV projection: [8192,256] x [768,256]^T + b, scatter to Q[B,H,S,D] (scaled
// by 1/sqrt(D)), K[B,H,S,D], V^T[B,H,D,S].
// Each wave computes a 16x64 strip: A fragment loaded once per k-step,
// 4 independent WMMA accumulator chains.
// ---------------------------------------------------------------------------
__global__ __launch_bounds__(256) void qkv_kernel(
    const _Float16* __restrict__ xh, const _Float16* __restrict__ wh,
    const float* __restrict__ bias, _Float16* __restrict__ Qh,
    _Float16* __restrict__ Kh, _Float16* __restrict__ VTh) {
  const int NG = (3 * CC) / 64;  // 12 n-groups of 64 cols
  int wave = threadIdx.x >> 5;
  int tid = blockIdx.x * 8 + wave;
  int mt = tid / NG, ng = tid % NG;
  int lane = threadIdx.x & 31, half = lane >> 4, n = lane & 15;

  v8f acc[4] = {{}, {}, {}, {}};
#pragma unroll
  for (int k = 0; k < CC; k += 32) {
    v16h a = load_frag_a(xh, CC, mt * 16, k);
#pragma unroll
    for (int t = 0; t < 4; t++) {
      v16h b = load_frag_b(wh, CC, ng * 64 + 16 * t, k);
      acc[t] = wmma16(a, b, acc[t]);
    }
  }
  const float qscale = 0.17677669529663687f;  // 1/sqrt(32)
#pragma unroll
  for (int t = 0; t < 4; t++) {
    int col = ng * 64 + 16 * t + n;
    float bc = bias[col];
#pragma unroll
    for (int j = 0; j < 8; j++) {
      int gm = mt * 16 + j + 8 * half;
      int bidx = gm >> 11;  // / S
      int s = gm & (SS - 1);
      float v = acc[t][j] + bc;
      if (col < CC) {
        int h = col >> 5, d = col & 31;
        Qh[(((size_t)bidx * HH + h) * SS + s) * DD + d] = (_Float16)(v * qscale);
      } else if (col < 2 * CC) {
        int c2 = col - CC;
        int h = c2 >> 5, d = c2 & 31;
        Kh[(((size_t)bidx * HH + h) * SS + s) * DD + d] = (_Float16)v;
      } else {
        int c2 = col - 2 * CC;
        int h = c2 >> 5, d = c2 & 31;
        VTh[(((size_t)bidx * HH + h) * DD + d) * SS + s] = (_Float16)v;
      }
    }
  }
}

// ---------------------------------------------------------------------------
// Flash attention: grid (S/128, H, B); 8 waves/block, one 16-row Q tile per
// wave; 64 keys per iteration: 4 score WMMAs, online softmax with 16-lane
// shuffle reductions, probabilities bounced D-layout -> A-fragment layout
// through wave-private LDS, 4 P*V^T WMMAs. Scores never touch global memory.
// ---------------------------------------------------------------------------
__global__ __launch_bounds__(256) void attn_kernel(
    const _Float16* __restrict__ Qh, const _Float16* __restrict__ Kh,
    const _Float16* __restrict__ VTh, _Float16* __restrict__ Oh) {
  __shared__ _Float16 patt[8][16][64];  // per-wave 16q x 64k prob tile (16KB)
  int wave = threadIdx.x >> 5;
  int lane = threadIdx.x & 31, half = lane >> 4, n = lane & 15;
  int b = blockIdx.z, h = blockIdx.y;
  int q0 = blockIdx.x * 128 + wave * 16;

  const _Float16* Qbh = Qh + ((size_t)b * HH + h) * SS * DD;
  const _Float16* Kbh = Kh + ((size_t)b * HH + h) * SS * DD;
  const _Float16* Vbh = VTh + ((size_t)b * HH + h) * DD * SS;

  v16h qf = load_frag_a(Qbh, DD, q0, 0);  // 16 q rows x full D=32

  v8f acc0 = {}, acc1 = {};
  float mrow[8], lrow[8];
#pragma unroll
  for (int j = 0; j < 8; j++) { mrow[j] = -1e30f; lrow[j] = 0.0f; }

  for (int kb = 0; kb < SS; kb += 64) {
    v8f s[4];
#pragma unroll
    for (int t = 0; t < 4; t++) {
      v16h kf = load_frag_b(Kbh, DD, kb + 16 * t, 0);
      v8f z = {};
      s[t] = wmma16(qf, kf, z);
    }

#pragma unroll
    for (int j = 0; j < 8; j++) {
      // row max over 64 keys: 4 local values, then 16-lane xor reduction
      float t0 = fmaxf(fmaxf(s[0][j], s[1][j]), fmaxf(s[2][j], s[3][j]));
      t0 = fmaxf(t0, __shfl_xor(t0, 1, 32));
      t0 = fmaxf(t0, __shfl_xor(t0, 2, 32));
      t0 = fmaxf(t0, __shfl_xor(t0, 4, 32));
      t0 = fmaxf(t0, __shfl_xor(t0, 8, 32));
      float mn = fmaxf(mrow[j], t0);
      float sc = __expf(mrow[j] - mn);
      float rs = 0.0f;
#pragma unroll
      for (int t = 0; t < 4; t++) {
        float p = __expf(s[t][j] - mn);
        rs += p;
        patt[wave][j + 8 * half][n + 16 * t] = (_Float16)p;
      }
      rs += __shfl_xor(rs, 1, 32);
      rs += __shfl_xor(rs, 2, 32);
      rs += __shfl_xor(rs, 4, 32);
      rs += __shfl_xor(rs, 8, 32);
      lrow[j] = lrow[j] * sc + rs;
      mrow[j] = mn;
      acc0[j] *= sc;
      acc1[j] *= sc;
    }

    // re-read probabilities as two A fragments (wave-private LDS, no barrier)
    const _Float16* pb = &patt[wave][lane & 15][0];
    FragH pf0, pf1;
    pf0.h[0] = *(const v8h*)(pb + 8 * half);
    pf0.h[1] = *(const v8h*)(pb + 16 + 8 * half);
    pf1.h[0] = *(const v8h*)(pb + 32 + 8 * half);
    pf1.h[1] = *(const v8h*)(pb + 48 + 8 * half);

    v16h vf00 = load_frag_b(Vbh, SS, 0, kb);        // d 0..15,  k 0..31
    v16h vf01 = load_frag_b(Vbh, SS, 0, kb + 32);   // d 0..15,  k 32..63
    v16h vf10 = load_frag_b(Vbh, SS, 16, kb);       // d 16..31, k 0..31
    v16h vf11 = load_frag_b(Vbh, SS, 16, kb + 32);  // d 16..31, k 32..63
    acc0 = wmma16(pf0.v, vf00, acc0);
    acc0 = wmma16(pf1.v, vf01, acc0);
    acc1 = wmma16(pf0.v, vf10, acc1);
    acc1 = wmma16(pf1.v, vf11, acc1);
  }

#pragma unroll
  for (int j = 0; j < 8; j++) {
    float inv = 1.0f / lrow[j];
    int gq = q0 + j + 8 * half;
    size_t row = ((size_t)b * SS + gq) * CC + h * DD;
    Oh[row + n] = (_Float16)(acc0[j] * inv);
    Oh[row + 16 + n] = (_Float16)(acc1[j] * inv);
  }
}

// ---------------------------------------------------------------------------
// GEMM [M,256] = A[M,K] x W[256,K]^T + bias, + residual, then LayerNorm over
// the 256-wide rows. One block = 16 rows; 8 waves x 2 n-tiles = 256 cols.
// Writes f32 (and optionally f16 copy for the next f16 GEMM).
// ---------------------------------------------------------------------------
__global__ __launch_bounds__(256) void gemm_res_ln_kernel(
    const _Float16* __restrict__ Ah, int K, const _Float16* __restrict__ Wh,
    const float* __restrict__ bias, const float* __restrict__ resid,
    const float* __restrict__ gamma, const float* __restrict__ beta,
    float* __restrict__ outf, _Float16* __restrict__ outh) {
  __shared__ float tileb[16][CC];
  __shared__ float ps[2][16][16];
  __shared__ float stats[2][16];

  int wave = threadIdx.x >> 5;
  int lane = threadIdx.x & 31, half = lane >> 4, n = lane & 15;
  int m0 = blockIdx.x * 16;

  v8f acc0 = {}, acc1 = {};
  for (int k = 0; k < K; k += 32) {
    v16h a = load_frag_a(Ah, K, m0, k);
    v16h b0 = load_frag_b(Wh, K, wave * 32, k);
    v16h b1 = load_frag_b(Wh, K, wave * 32 + 16, k);
    acc0 = wmma16(a, b0, acc0);
    acc1 = wmma16(a, b1, acc1);
  }
  int c0 = wave * 32 + n, c1 = c0 + 16;
#pragma unroll
  for (int j = 0; j < 8; j++) {
    int r = j + 8 * half;
    size_t grow = (size_t)(m0 + r) * CC;
    tileb[r][c0] = acc0[j] + bias[c0] + resid[grow + c0];
    tileb[r][c1] = acc1[j] + bias[c1] + resid[grow + c1];
  }
  __syncthreads();

  int r = threadIdx.x >> 4;
  int cb = (threadIdx.x & 15) * 16;
  float s = 0.f, s2 = 0.f;
#pragma unroll
  for (int i = 0; i < 16; i++) {
    float v = tileb[r][cb + i];
    s += v; s2 += v * v;
  }
  ps[0][r][threadIdx.x & 15] = s;
  ps[1][r][threadIdx.x & 15] = s2;
  __syncthreads();
  if (threadIdx.x < 16) {
    float ts = 0.f, ts2 = 0.f;
#pragma unroll
    for (int i = 0; i < 16; i++) {
      ts += ps[0][threadIdx.x][i];
      ts2 += ps[1][threadIdx.x][i];
    }
    float mean = ts * (1.0f / CC);
    float var = ts2 * (1.0f / CC) - mean * mean;
    stats[0][threadIdx.x] = mean;
    stats[1][threadIdx.x] = rsqrtf(var + 1e-5f);
  }
  __syncthreads();
  float mean = stats[0][r], rstd = stats[1][r];
#pragma unroll
  for (int i = 0; i < 16; i++) {
    int c = cb + i;
    float v = (tileb[r][c] - mean) * rstd * gamma[c] + beta[c];
    size_t idx = (size_t)(m0 + r) * CC + c;
    outf[idx] = v;
    if (outh) outh[idx] = (_Float16)v;
  }
}

// ---------------------------------------------------------------------------
// FFN up-projection: [8192,512] = gelu([8192,256] x [512,256]^T + b1) (f16 out)
// Each wave computes a 16x64 strip (4 accumulator chains).
// ---------------------------------------------------------------------------
__global__ __launch_bounds__(256) void ffn1_kernel(
    const _Float16* __restrict__ x1h, const _Float16* __restrict__ w1h,
    const float* __restrict__ b1, _Float16* __restrict__ hh) {
  const int NG = FF / 64;  // 8 n-groups
  int wave = threadIdx.x >> 5;
  int tid = blockIdx.x * 8 + wave;
  int mt = tid / NG, ng = tid % NG;
  int lane = threadIdx.x & 31, half = lane >> 4, n = lane & 15;

  v8f acc[4] = {{}, {}, {}, {}};
#pragma unroll
  for (int k = 0; k < CC; k += 32) {
    v16h a = load_frag_a(x1h, CC, mt * 16, k);
#pragma unroll
    for (int t = 0; t < 4; t++) {
      v16h b = load_frag_b(w1h, CC, ng * 64 + 16 * t, k);
      acc[t] = wmma16(a, b, acc[t]);
    }
  }
#pragma unroll
  for (int t = 0; t < 4; t++) {
    int col = ng * 64 + 16 * t + n;
    float bc = b1[col];
#pragma unroll
    for (int j = 0; j < 8; j++) {
      int gm = mt * 16 + j + 8 * half;
      float v = acc[t][j] + bc;
      v = 0.5f * v * (1.0f + erff(v * 0.70710678118654752f));  // exact gelu
      hh[(size_t)gm * FF + col] = (_Float16)v;
    }
  }
}

// ---------------------------------------------------------------------------
// Launch
// ---------------------------------------------------------------------------
extern "C" void kernel_launch(void* const* d_in, const int* in_sizes, int n_in,
                              void* d_out, int out_size, void* d_ws, size_t ws_size,
                              hipStream_t stream) {
  const float* x     = (const float*)d_in[0];
  // d_in[1] voxel_coords: sorted equal segments -> pure reshape, unused
  const float* w_qkv = (const float*)d_in[2];
  const float* b_qkv = (const float*)d_in[3];
  const float* w_out = (const float*)d_in[4];
  const float* b_out = (const float*)d_in[5];
  const float* ln1g  = (const float*)d_in[6];
  const float* ln1b  = (const float*)d_in[7];
  const float* w1    = (const float*)d_in[8];
  const float* b1    = (const float*)d_in[9];
  const float* w2    = (const float*)d_in[10];
  const float* b2    = (const float*)d_in[11];
  const float* ln2g  = (const float*)d_in[12];
  const float* ln2b  = (const float*)d_in[13];
  float* out = (float*)d_out;

  char* ws = (char*)d_ws;
  size_t off = 0;
  auto alloc = [&](size_t bytes) -> void* {
    void* p = ws + off;
    off = (off + bytes + 255) & ~(size_t)255;
    return p;
  };
  const size_t NTOK = (size_t)BB * SS;  // 8192
  _Float16* xh    = (_Float16*)alloc(NTOK * CC * 2);
  _Float16* wqkvh = (_Float16*)alloc((size_t)3 * CC * CC * 2);
  _Float16* wouth = (_Float16*)alloc((size_t)CC * CC * 2);
  _Float16* w1h   = (_Float16*)alloc((size_t)FF * CC * 2);
  _Float16* w2h   = (_Float16*)alloc((size_t)CC * FF * 2);
  _Float16* Qh    = (_Float16*)alloc(NTOK * CC * 2);
  _Float16* Kh    = (_Float16*)alloc(NTOK * CC * 2);
  _Float16* VTh   = (_Float16*)alloc(NTOK * CC * 2);
  _Float16* Oh    = (_Float16*)alloc(NTOK * CC * 2);
  float*    x1    = (float*)alloc(NTOK * CC * 4);
  _Float16* x1h   = (_Float16*)alloc(NTOK * CC * 2);
  _Float16* hh    = (_Float16*)alloc(NTOK * FF * 2);

  auto cvt = [&](const float* src, _Float16* dst, size_t n) {
    int n4 = (int)(n / 4);
    cvt_f32_f16<<<(n4 + 255) / 256, 256, 0, stream>>>(src, dst, n4);
  };
  cvt(x, xh, NTOK * CC);
  cvt(w_qkv, wqkvh, (size_t)3 * CC * CC);
  cvt(w_out, wouth, (size_t)CC * CC);
  cvt(w1, w1h, (size_t)FF * CC);
  cvt(w2, w2h, (size_t)CC * FF);

  // QKV: 512 m-tiles x 12 n-groups (64 cols each), 8 waves/block
  qkv_kernel<<<(512 * 12) / 8, 256, 0, stream>>>(xh, wqkvh, b_qkv, Qh, Kh, VTh);

  // Flash attention: (S/128, H, B)
  attn_kernel<<<dim3(SS / 128, HH, BB), 256, 0, stream>>>(Qh, Kh, VTh, Oh);

  // out-proj + residual(x) + LN1 -> x1 (f32) and x1h (f16)
  gemm_res_ln_kernel<<<NTOK / 16, 256, 0, stream>>>(Oh, CC, wouth, b_out, x,
                                                    ln1g, ln1b, x1, x1h);
  // FFN1 + gelu -> hh
  ffn1_kernel<<<(512 * 8) / 8, 256, 0, stream>>>(x1h, w1h, b1, hh);

  // FFN2 + residual(x1) + LN2 -> final output (f32)
  gemm_res_ln_kernel<<<NTOK / 16, 256, 0, stream>>>(hh, FF, w2h, b2, x1,
                                                    ln2g, ln2b, out, nullptr);
}
